// SparseLinear_44427141710512
// MI455X (gfx1250) — matrix operand order repaired and checked by reading, not compiled, not run on hardware
//
#include <hip/hip_runtime.h>

// ---------------------------------------------------------------------------
// y[M,N] = x[M,K] * W[K,N] + bias[N],  M=8192, N=K=4096, f32 in / f32 out.
// bf16x3 split-precision GEMM on the CDNA5 WMMA bf16 pipe (16x16x32),
// f32 accumulation. Workgroup tile 128x128, TK=32, 8 waves (wave32),
// wave tile 32x64 -> 2x4 accumulator tiles, 3 WMMA (hi*hi, hi*lo, lo*hi) each.
// ---------------------------------------------------------------------------

#define TM 128
#define TN 128
#define TK 32
#define LDA 40   // padded bf16 elements per A row (32 data + 8 pad = 80B)
#define LDB 40   // padded bf16 elements per Bt row (n-major, 32 K + 8 pad)

typedef __bf16 bf16_t;
typedef __attribute__((ext_vector_type(16))) __bf16 v16bf;
typedef __attribute__((ext_vector_type(8)))  __bf16 v8bf;
typedef __attribute__((ext_vector_type(4)))  __bf16 v4bf;
typedef __attribute__((ext_vector_type(8)))  float  v8f;
typedef __attribute__((ext_vector_type(4)))  float  v4f;

union Frag16 { v16bf v; v8bf h[2]; };

// A fragment (16x32 bf16), ISA 7.12.2 layout:
//   lane half h = lane>>4, row m = lane&15
//   elements 0..7  = K in [8h, 8h+8)      (VGPR0..3)
//   elements 8..15 = K in [16+8h, 24+8h)  (VGPR4..7)
__device__ __forceinline__ v16bf load_frag_a(const bf16_t* base, int row, int h) {
  Frag16 f;
  const bf16_t* p = base + row * LDA + 8 * h;
  f.h[0] = *(const v8bf*)(p);
  f.h[1] = *(const v8bf*)(p + 16);
  return f.v;
}

// B fragment (32x16 bf16) from column-major (n-major) LDS tile:
//   lane half h = lane>>4, col n = lane&15
//   elements 0..7 = K in [8h, 8h+8), elements 8..15 = K in [16+8h, 24+8h)
__device__ __forceinline__ v16bf load_frag_b(const bf16_t* base, int nrow, int h) {
  Frag16 f;
  const bf16_t* p = base + nrow * LDB + 8 * h;
  f.h[0] = *(const v8bf*)(p);
  f.h[1] = *(const v8bf*)(p + 16);
  return f.v;
}

__global__ void __launch_bounds__(256)
sparse_linear_wmma_bf16x3(const float* __restrict__ X,
                          const float* __restrict__ Wm,
                          const float* __restrict__ bias,
                          float* __restrict__ Y,
                          int M, int N, int K)
{
  __shared__ __align__(16) bf16_t sAhi[TM * LDA];
  __shared__ __align__(16) bf16_t sAlo[TM * LDA];
  __shared__ __align__(16) bf16_t sBhi[TN * LDB];   // transposed: [n][k]
  __shared__ __align__(16) bf16_t sBlo[TN * LDB];

  const int tid  = threadIdx.x;
  const int wid  = tid >> 5;      // wave id 0..7
  const int lane = tid & 31;
  const int lh   = lane >> 4;     // lane half (0/1)
  const int lm   = lane & 15;

  const int mWG = blockIdx.y * TM;
  const int nWG = blockIdx.x * TN;

  const int wM = (wid >> 1) * 32;  // 4 wave rows of 32
  const int wN = (wid & 1) * 64;   // 2 wave cols of 64

  // --- global load mapping ---------------------------------------------
  // A tile: 128 rows x 32 K.  thread -> row = tid/2, K-half = (tid&1)*16
  const int arow = tid >> 1;
  const int acol = (tid & 1) * 16;
  const float* gA = X + (size_t)(mWG + arow) * K + acol;

  // B tile: 32 K x 128 n as 4x4 micro-tiles for the register transpose.
  //   cg = tid % 32 (n group), rg = tid / 32 (K group)
  const int bcg = tid & 31;
  const int brg = tid >> 5;
  const float* gB = Wm + (size_t)(brg * 4) * N + nWG + bcg * 4;

  v4f ra[4], rb[4];
#pragma unroll
  for (int i = 0; i < 4; ++i) ra[i] = *(const v4f*)(gA + 4 * i);
#pragma unroll
  for (int i = 0; i < 4; ++i) rb[i] = *(const v4f*)(gB + (size_t)i * N);

  v8f acc[2][4];
#pragma unroll
  for (int mt = 0; mt < 2; ++mt)
#pragma unroll
    for (int nt = 0; nt < 4; ++nt)
#pragma unroll
      for (int r = 0; r < 8; ++r) acc[mt][nt][r] = 0.0f;

  const int NK = K / TK;  // 128 K-steps
  for (int kt = 0; kt < NK; ++kt) {
    // ---- split f32 -> bf16 hi/lo and stage into LDS -------------------
#pragma unroll
    for (int i = 0; i < 4; ++i) {
      v4bf h4, l4;
#pragma unroll
      for (int j = 0; j < 4; ++j) {
        float v  = ra[i][j];
        __bf16 hb = (__bf16)v;                 // RNE
        h4[j] = hb;
        l4[j] = (__bf16)(v - (float)hb);       // residual
      }
      *(v4bf*)(&sAhi[arow * LDA + acol + 4 * i]) = h4;
      *(v4bf*)(&sAlo[arow * LDA + acol + 4 * i]) = l4;
    }
    // register-transpose the 4x4 W micro-tile -> n-major LDS
#pragma unroll
    for (int j = 0; j < 4; ++j) {
      v4bf h4, l4;
#pragma unroll
      for (int i = 0; i < 4; ++i) {
        float v  = rb[i][j];
        __bf16 hb = (__bf16)v;
        h4[i] = hb;
        l4[i] = (__bf16)(v - (float)hb);
      }
      *(v4bf*)(&sBhi[(bcg * 4 + j) * LDB + brg * 4]) = h4;
      *(v4bf*)(&sBlo[(bcg * 4 + j) * LDB + brg * 4]) = l4;
    }
    __syncthreads();

    // ---- prefetch next K-tile into registers (overlaps WMMA) ----------
    if (kt + 1 < NK) {
      const float* pA = gA + (size_t)(kt + 1) * TK;
#pragma unroll
      for (int i = 0; i < 4; ++i) ra[i] = *(const v4f*)(pA + 4 * i);
      const float* pB = gB + (size_t)(kt + 1) * TK * N;
#pragma unroll
      for (int i = 0; i < 4; ++i) rb[i] = *(const v4f*)(pB + (size_t)i * N);
    }

    // ---- WMMA: 2 (M) x 4 (N) tiles x 3 split-products -----------------
    v16bf ah[2], al[2];
#pragma unroll
    for (int mt = 0; mt < 2; ++mt) {
      int row = wM + mt * 16 + lm;
      ah[mt] = load_frag_a(sAhi, row, lh);
      al[mt] = load_frag_a(sAlo, row, lh);
    }
#pragma unroll
    for (int nt = 0; nt < 4; ++nt) {
      int nrow = wN + nt * 16 + lm;
      v16bf bh = load_frag_b(sBhi, nrow, lh);
      v16bf bl = load_frag_b(sBlo, nrow, lh);
#pragma unroll
      for (int mt = 0; mt < 2; ++mt) {
        acc[mt][nt] = __builtin_amdgcn_wmma_f32_16x16x32_bf16(
            false, ah[mt], false, bh, (short)0, acc[mt][nt], false, false);
        acc[mt][nt] = __builtin_amdgcn_wmma_f32_16x16x32_bf16(
            false, ah[mt], false, bl, (short)0, acc[mt][nt], false, false);
        acc[mt][nt] = __builtin_amdgcn_wmma_f32_16x16x32_bf16(
            false, al[mt], false, bh, (short)0, acc[mt][nt], false, false);
      }
    }
    __syncthreads();
  }

  // ---- epilogue: fused bias add, C/D layout M = r + 8*lh, N = lm ------
#pragma unroll
  for (int nt = 0; nt < 4; ++nt) {
    int gcol = nWG + wN + nt * 16 + lm;
    float bb = bias[gcol];
#pragma unroll
    for (int mt = 0; mt < 2; ++mt) {
      int gr0 = mWG + wM + mt * 16 + 8 * lh;
#pragma unroll
      for (int r = 0; r < 8; ++r) {
        Y[(size_t)(gr0 + r) * N + gcol] = acc[mt][nt][r] + bb;
      }
    }
  }
}

extern "C" void kernel_launch(void* const* d_in, const int* in_sizes, int n_in,
                              void* d_out, int out_size, void* d_ws, size_t ws_size,
                              hipStream_t stream) {
  const float* x    = (const float*)d_in[0];
  const float* W    = (const float*)d_in[1];
  const float* bias = (const float*)d_in[2];
  float* out = (float*)d_out;

  const int D = 4096;
  const int M = (n_in > 0 && in_sizes) ? (in_sizes[0] / D) : 8192;  // B*S = 8192
  const int N = D, K = D;

  dim3 grid(N / TN, M / TM);  // 32 x 64 workgroups
  sparse_linear_wmma_bf16x3<<<grid, dim3(256), 0, stream>>>(x, W, bias, out, M, N, K);
}
